// GraphformerLayer_15479062135324
// MI455X (gfx1250) — matrix operand order repaired and checked by reading, not compiled, not run on hardware
//
#include <hip/hip_runtime.h>

// Problem constants (from reference)
#define NROWS    65536
#define IN_DIM   256
#define OUT_DIM  64
#define HEADS    8
#define PROJ_DIM 64
#define HD       512          // HEADS*OUT_DIM
#define KEPS     1e-3f
#define ZEPS     1e-6f
#define NORMC    0.35355339059327373f   // 64^-0.25

#define ROWS_PER_BLK 128
#define THREADS      256      // 8 wave32

typedef __bf16 bf16_t;
typedef __attribute__((ext_vector_type(16))) __bf16 v16bf;
typedef __attribute__((ext_vector_type(8)))  float  v8f;

#define WMMA_BF16(a, b, c) \
  __builtin_amdgcn_wmma_f32_16x16x32_bf16(false, (a), false, (b), (short)0, (c), false, false)

union Pack4 { uint2 u; bf16_t h[4]; };
union Pack8 { uint4 u; bf16_t h[8]; };

static __device__ inline v8f zf8() {
  v8f a;
#pragma unroll
  for (int i = 0; i < 8; ++i) a[i] = 0.0f;
  return a;
}

// A-fragment (16x32 bf16, M x K) from row-major [rows][stride] bf16 in LDS.
// lane 0-15: row=lane, K chunks {0..7},{16..23}; lane 16-31: row=lane-16, {8..15},{24..31}
static __device__ inline v16bf ld_afrag(const bf16_t* base, int row0, int stride, int k0) {
  const unsigned lane = threadIdx.x & 31u;
  const unsigned m  = lane & 15u;
  const unsigned hi = lane >> 4;
  const bf16_t* p = base + (row0 + (int)m) * stride + k0 + (int)(hi * 8);
  union { v16bf v; uint4 q[2]; } r;
  r.q[0] = *(const uint4*)(p);
  r.q[1] = *(const uint4*)(p + 16);
  return r.v;
}

// B-fragment (32x16 bf16, K x N) from "n-major" [n][stride] bf16 in LDS (contiguous in k).
// lane 0-15: col n=lane, K chunks {0..7},{16..23}; lane 16-31: col n=lane-16, {8..15},{24..31}
static __device__ inline v16bf ld_bfrag(const bf16_t* base, int n0, int stride, int k0) {
  const unsigned lane = threadIdx.x & 31u;
  const unsigned n  = lane & 15u;
  const unsigned hi = lane >> 4;
  const bf16_t* p = base + (n0 + (int)n) * stride + k0 + (int)(hi * 8);
  union { v16bf v; uint4 q[2]; } r;
  r.q[0] = *(const uint4*)(p);
  r.q[1] = *(const uint4*)(p + 16);
  return r.v;
}

// One k-step of a 16x64 output strip: group all 4 B-frag loads before the
// 4 WMMAs so only one s_wait_dscnt gates four back-to-back matrix ops.
static __device__ inline void gemm4(v8f acc[4], v16bf a,
                                    const bf16_t* bbase, int stride, int k0) {
  v16bf b0 = ld_bfrag(bbase,  0, stride, k0);
  v16bf b1 = ld_bfrag(bbase, 16, stride, k0);
  v16bf b2 = ld_bfrag(bbase, 32, stride, k0);
  v16bf b3 = ld_bfrag(bbase, 48, stride, k0);
  acc[0] = WMMA_BF16(a, b0, acc[0]);
  acc[1] = WMMA_BF16(a, b1, acc[1]);
  acc[2] = WMMA_BF16(a, b2, acc[2]);
  acc[3] = WMMA_BF16(a, b3, acc[3]);
}

// Cooperative fp32 -> bf16 staging (nelem multiple of 4*THREADS)
static __device__ inline void stage_bf16(const float* __restrict__ g, bf16_t* s, int nelem) {
  for (int i = (int)threadIdx.x * 4; i < nelem; i += THREADS * 4) {
    float4 f = *(const float4*)(g + i);
    Pack4 r;
    r.h[0] = (bf16_t)f.x; r.h[1] = (bf16_t)f.y;
    r.h[2] = (bf16_t)f.z; r.h[3] = (bf16_t)f.w;
    *(uint2*)(s + i) = r.u;
  }
}

__global__ void zero_ws_kernel(float* __restrict__ p, int n) {
  int i = blockIdx.x * blockDim.x + threadIdx.x;
  if (i < n) p[i] = 0.0f;
}

// ---------------------------------------------------------------------------
// Pass 1: K-side. Accumulate KV[h][m][d] and Ksum[h][m] over all rows.
// ---------------------------------------------------------------------------
__global__ __launch_bounds__(THREADS) void pass1_kv_kernel(
    const float* __restrict__ x,  const float* __restrict__ Wk,
    const float* __restrict__ Wv, const float* __restrict__ P,
    float* __restrict__ KV, float* __restrict__ Ksum)
{
  __shared__ bf16_t s_x[ROWS_PER_BLK * IN_DIM];        // 64 KB, bf16 x tile
  __shared__ bf16_t s_w[OUT_DIM * IN_DIM];             // 32 KB, Wk_h / Wv_h
  __shared__ bf16_t s_p[PROJ_DIM * OUT_DIM];           //  8 KB, P row-major
  __shared__ bf16_t s_kpT[PROJ_DIM * ROWS_PER_BLK];    // 16 KB, Kp^T [m][row]
  __shared__ bf16_t s_vT[OUT_DIM * ROWS_PER_BLK];      // 16 KB, V^T  [d][row]
  __shared__ bf16_t s_t1[8 * 16 * 64];                 // 16 KB, per-wave Kn slab

  const int tid = threadIdx.x;
  const int wid = tid >> 5;
  const unsigned lane = tid & 31u;
  const unsigned nlo  = lane & 15u;
  const unsigned mhi  = (lane >> 4) * 8;
  const int wrow = wid * 16;
  const long grow = (long)blockIdx.x * ROWS_PER_BLK;

  stage_bf16(x + grow * IN_DIM, s_x, ROWS_PER_BLK * IN_DIM);
  stage_bf16(P, s_p, PROJ_DIM * OUT_DIM);
  __syncthreads();

  // x A-fragments: loop-invariant across heads and across K/V GEMMs.
  v16bf xa[8];
#pragma unroll
  for (int kk = 0; kk < 8; ++kk) xa[kk] = ld_afrag(s_x, wrow, IN_DIM, kk * 32);

  bf16_t* my1 = s_t1 + wid * (16 * 64);

  for (int h = 0; h < HEADS; ++h) {
    __syncthreads();
    stage_bf16(Wk + (long)h * OUT_DIM * IN_DIM, s_w, OUT_DIM * IN_DIM);
    __syncthreads();

    // K_h[16,64] = x_slab @ Wk_h^T
    v8f ak[4];
#pragma unroll
    for (int nt = 0; nt < 4; ++nt) ak[nt] = zf8();
#pragma unroll
    for (int kk = 0; kk < 8; ++kk) gemm4(ak, xa[kk], s_w, IN_DIM, kk * 32);

    // Kn = norm*K -> per-wave slab (row-major)
#pragma unroll
    for (int nt = 0; nt < 4; ++nt)
#pragma unroll
      for (int v = 0; v < 8; ++v)
        my1[(mhi + v) * 64 + nt * 16 + nlo] = (bf16_t)(ak[nt][v] * NORMC);

    // Kp = relu(Kn @ P^T) + eps
    v8f ap[4];
#pragma unroll
    for (int nt = 0; nt < 4; ++nt) ap[nt] = zf8();
#pragma unroll
    for (int kk = 0; kk < 2; ++kk)
      gemm4(ap, ld_afrag(my1, 0, 64, kk * 32), s_p, OUT_DIM, kk * 32);

    // store transposed: s_kpT[m][row] (8 contiguous rows per lane -> b128 stores)
#pragma unroll
    for (int nt = 0; nt < 4; ++nt) {
      Pack8 r;
#pragma unroll
      for (int v = 0; v < 8; ++v) r.h[v] = (bf16_t)(fmaxf(ap[nt][v], 0.0f) + KEPS);
      *(uint4*)(s_kpT + (nt * 16 + nlo) * ROWS_PER_BLK + wrow + mhi) = r.u;
    }
    __syncthreads();

    // Ksum partial (column sums over this block's 128 rows)
    if (tid < PROJ_DIM) {
      float s = 0.0f;
      for (int r = 0; r < ROWS_PER_BLK; ++r) s += (float)s_kpT[tid * ROWS_PER_BLK + r];
      atomicAdd(Ksum + h * PROJ_DIM + tid, s);
    }
    stage_bf16(Wv + (long)h * OUT_DIM * IN_DIM, s_w, OUT_DIM * IN_DIM);
    __syncthreads();

    // V_h[16,64] = x_slab @ Wv_h^T
    v8f av[4];
#pragma unroll
    for (int nt = 0; nt < 4; ++nt) av[nt] = zf8();
#pragma unroll
    for (int kk = 0; kk < 8; ++kk) gemm4(av, xa[kk], s_w, IN_DIM, kk * 32);

    // store transposed: s_vT[d][row]
#pragma unroll
    for (int nt = 0; nt < 4; ++nt) {
      Pack8 r;
#pragma unroll
      for (int v = 0; v < 8; ++v) r.h[v] = (bf16_t)av[nt][v];
      *(uint4*)(s_vT + (nt * 16 + nlo) * ROWS_PER_BLK + wrow + mhi) = r.u;
    }
    __syncthreads();

    // KV partial: Kp^T[64,128] @ V[128,64]; 16 output tiles, 2 per wave.
    // Group all 8 fragment loads per tile -> one wait, 4 chained WMMAs.
#pragma unroll
    for (int t = 0; t < 2; ++t) {
      const int tile = wid * 2 + t;
      const int mt = tile >> 2, dt = tile & 3;
      v16bf a0 = ld_afrag(s_kpT, mt * 16, ROWS_PER_BLK, 0);
      v16bf a1 = ld_afrag(s_kpT, mt * 16, ROWS_PER_BLK, 32);
      v16bf a2 = ld_afrag(s_kpT, mt * 16, ROWS_PER_BLK, 64);
      v16bf a3 = ld_afrag(s_kpT, mt * 16, ROWS_PER_BLK, 96);
      v16bf b0 = ld_bfrag(s_vT,  dt * 16, ROWS_PER_BLK, 0);
      v16bf b1 = ld_bfrag(s_vT,  dt * 16, ROWS_PER_BLK, 32);
      v16bf b2 = ld_bfrag(s_vT,  dt * 16, ROWS_PER_BLK, 64);
      v16bf b3 = ld_bfrag(s_vT,  dt * 16, ROWS_PER_BLK, 96);
      v8f acc = zf8();
      acc = WMMA_BF16(a0, b0, acc);
      acc = WMMA_BF16(a1, b1, acc);
      acc = WMMA_BF16(a2, b2, acc);
      acc = WMMA_BF16(a3, b3, acc);
      float* dst = KV + h * (PROJ_DIM * OUT_DIM);
#pragma unroll
      for (int v = 0; v < 8; ++v)
        atomicAdd(dst + (mt * 16 + mhi + v) * OUT_DIM + dt * 16 + nlo, acc[v]);
    }
  }
}

// ---------------------------------------------------------------------------
// Pass 2: Q-side. out = ((relu(normQ@P^T)+eps) @ KV) * z, y = out @ Wo^T + bo.
// ---------------------------------------------------------------------------
__global__ __launch_bounds__(THREADS) void pass2_q_kernel(
    const float* __restrict__ x,  const float* __restrict__ Wq,
    const float* __restrict__ P,  const float* __restrict__ Wo,
    const float* __restrict__ bo, const float* __restrict__ KV,
    const float* __restrict__ Ksum, float* __restrict__ out)
{
  __shared__ bf16_t s_x[ROWS_PER_BLK * IN_DIM];   // 64 KB
  __shared__ bf16_t s_w[OUT_DIM * IN_DIM];        // 32 KB Wq_h
  __shared__ bf16_t s_p[PROJ_DIM * OUT_DIM];      //  8 KB
  __shared__ bf16_t s_kvT[OUT_DIM * PROJ_DIM];    //  8 KB KV_h^T [d][m]
  __shared__ bf16_t s_wo[OUT_DIM * OUT_DIM];      //  8 KB Wo slice [n][k]
  __shared__ bf16_t s_t1[8 * 16 * 64];            // 16 KB per-wave Qn/Qp slab
  __shared__ bf16_t s_t2[8 * 16 * 64];            // 16 KB per-wave out slab
  __shared__ float  s_ks[HD];                     //  2 KB Ksum
  __shared__ float  s_z[ROWS_PER_BLK];            // 512 B

  const int tid = threadIdx.x;
  const int wid = tid >> 5;
  const unsigned lane = tid & 31u;
  const unsigned nlo  = lane & 15u;
  const unsigned mhi  = (lane >> 4) * 8;
  const int wrow = wid * 16;
  const long grow = (long)blockIdx.x * ROWS_PER_BLK;

  stage_bf16(x + grow * IN_DIM, s_x, ROWS_PER_BLK * IN_DIM);
  stage_bf16(P, s_p, PROJ_DIM * OUT_DIM);
  s_ks[tid] = Ksum[tid];
  s_ks[tid + 256] = Ksum[tid + 256];
  __syncthreads();

  // x A-fragments: loop-invariant across heads.
  v16bf xa[8];
#pragma unroll
  for (int kk = 0; kk < 8; ++kk) xa[kk] = ld_afrag(s_x, wrow, IN_DIM, kk * 32);

  bf16_t* my1 = s_t1 + wid * (16 * 64);
  bf16_t* my2 = s_t2 + wid * (16 * 64);

  v8f accY[4];
#pragma unroll
  for (int nt = 0; nt < 4; ++nt) accY[nt] = zf8();

  for (int h = 0; h < HEADS; ++h) {
    __syncthreads();
    stage_bf16(Wq + (long)h * OUT_DIM * IN_DIM, s_w, OUT_DIM * IN_DIM);
    // KV_h transposed -> bf16 LDS: s_kvT[d][m]  (global reads coalesced in d)
    for (int i = tid; i < OUT_DIM * PROJ_DIM; i += THREADS) {
      int m = i >> 6, d = i & 63;
      s_kvT[d * PROJ_DIM + m] = (bf16_t)KV[h * (PROJ_DIM * OUT_DIM) + i];
    }
    // Wo slice: s_wo[n][k] = Wo[n][h*64+k]
    for (int i = tid * 4; i < OUT_DIM * OUT_DIM; i += THREADS * 4) {
      int n = i >> 6, k = i & 63;
      float4 f = *(const float4*)(Wo + (long)n * HD + h * OUT_DIM + k);
      Pack4 r;
      r.h[0] = (bf16_t)f.x; r.h[1] = (bf16_t)f.y;
      r.h[2] = (bf16_t)f.z; r.h[3] = (bf16_t)f.w;
      *(uint2*)(s_wo + i) = r.u;
    }
    __syncthreads();

    // Q_h[16,64] = x_slab @ Wq_h^T
    v8f aq[4];
#pragma unroll
    for (int nt = 0; nt < 4; ++nt) aq[nt] = zf8();
#pragma unroll
    for (int kk = 0; kk < 8; ++kk) gemm4(aq, xa[kk], s_w, IN_DIM, kk * 32);

    // Qn = norm*Q -> slab
#pragma unroll
    for (int nt = 0; nt < 4; ++nt)
#pragma unroll
      for (int v = 0; v < 8; ++v)
        my1[(mhi + v) * 64 + nt * 16 + nlo] = (bf16_t)(aq[nt][v] * NORMC);

    // Qp = relu(Qn @ P^T) + eps
    v8f ap[4];
#pragma unroll
    for (int nt = 0; nt < 4; ++nt) ap[nt] = zf8();
#pragma unroll
    for (int kk = 0; kk < 2; ++kk)
      gemm4(ap, ld_afrag(my1, 0, 64, kk * 32), s_p, OUT_DIM, kk * 32);
#pragma unroll
    for (int nt = 0; nt < 4; ++nt)
#pragma unroll
      for (int v = 0; v < 8; ++v)
        my1[(mhi + v) * 64 + nt * 16 + nlo] = (bf16_t)(fmaxf(ap[nt][v], 0.0f) + KEPS);

    // z = 1/(Qp . Ksum_h + eps), per-row (lanes 0-15, one row each)
    if (lane < 16) {
      float acc = 0.0f;
      for (int m = 0; m < PROJ_DIM; ++m)
        acc += (float)my1[(int)lane * 64 + m] * s_ks[h * PROJ_DIM + m];
      s_z[wid * 16 + (int)lane] = 1.0f / (acc + ZEPS);
    }

    // out_h = Qp @ KV_h
    v8f ao[4];
#pragma unroll
    for (int nt = 0; nt < 4; ++nt) ao[nt] = zf8();
#pragma unroll
    for (int kk = 0; kk < 2; ++kk)
      gemm4(ao, ld_afrag(my1, 0, 64, kk * 32), s_kvT, PROJ_DIM, kk * 32);

    float zv[8];
#pragma unroll
    for (int v = 0; v < 8; ++v) zv[v] = s_z[wid * 16 + (int)mhi + v];
#pragma unroll
    for (int nt = 0; nt < 4; ++nt)
#pragma unroll
      for (int v = 0; v < 8; ++v)
        my2[(mhi + v) * 64 + nt * 16 + nlo] = (bf16_t)(ao[nt][v] * zv[v]);

    // y += out_h @ Wo_h^T
#pragma unroll
    for (int kk = 0; kk < 2; ++kk)
      gemm4(accY, ld_afrag(my2, 0, 64, kk * 32), s_wo, OUT_DIM, kk * 32);
  }

  // epilogue: + bo, store fp32
#pragma unroll
  for (int nt = 0; nt < 4; ++nt) {
    float b = bo[nt * 16 + nlo];
#pragma unroll
    for (int v = 0; v < 8; ++v)
      out[(grow + wrow + (long)mhi + v) * OUT_DIM + nt * 16 + nlo] = accY[nt][v] + b;
  }
}

// ---------------------------------------------------------------------------
extern "C" void kernel_launch(void* const* d_in, const int* in_sizes, int n_in,
                              void* d_out, int out_size, void* d_ws, size_t ws_size,
                              hipStream_t stream) {
  (void)in_sizes; (void)n_in; (void)out_size; (void)ws_size;
  const float* x   = (const float*)d_in[0];
  const float* Wq  = (const float*)d_in[1];
  const float* Wk  = (const float*)d_in[2];
  const float* Wv  = (const float*)d_in[3];
  const float* Wo  = (const float*)d_in[4];
  const float* bo  = (const float*)d_in[5];
  const float* P   = (const float*)d_in[6];
  float* out  = (float*)d_out;
  float* KV   = (float*)d_ws;                        // [8][64][64]
  float* Ksum = KV + HEADS * PROJ_DIM * OUT_DIM;     // [8][64]

  const int nacc = HEADS * PROJ_DIM * OUT_DIM + HEADS * PROJ_DIM;  // 33280
  zero_ws_kernel<<<(nacc + THREADS - 1) / THREADS, THREADS, 0, stream>>>(KV, nacc);

  const int nblk = NROWS / ROWS_PER_BLK;  // 512
  pass1_kv_kernel<<<nblk, THREADS, 0, stream>>>(x, Wk, Wv, P, KV, Ksum);
  pass2_q_kernel <<<nblk, THREADS, 0, stream>>>(x, Wq, P, Wo, bo, KV, Ksum, out);
}
